// SelfAttention_19069654794581
// MI455X (gfx1250) — compile-verified, matrix operand order checked by reading, not run on hardware
//
#include <hip/hip_runtime.h>
#include <hip/hip_bf16.h>

// ---------------------------------------------------------------------------
// Problem constants (from reference)
// ---------------------------------------------------------------------------
#define DIM      512
#define HEADS    8
#define HDIM     64          // DIM / HEADS
#define BATCH    4
#define SEQ      8192
#define M_TOT    (BATCH * SEQ)     // 32768 rows of x (flattened [B,N])
#define J_TOT    (3 * DIM)         // 1536 qkv output columns
#define OUT_CHUNK ((size_t)BATCH * SEQ * DIM)  // 16,777,216 floats per output tensor

// GEMM tiling
#define MBLK 128
#define JBLK 128
#define KBLK 32
#define LDS_STRIDE 40            // bf16 elems per LDS row (80 B, 16B-aligned, conflict-free)

typedef __attribute__((ext_vector_type(16))) __bf16 v16bf;
typedef __attribute__((ext_vector_type(8)))  __bf16 v8bf;
typedef __attribute__((ext_vector_type(8)))  float  v8f;
typedef __attribute__((ext_vector_type(4)))  float  v4f;

// ---------------------------------------------------------------------------
// Kernel 1: qkv = x @ qkv_w^T + qkv_b, scattered into q/k/v [N,H,B,Hd] fp32.
//   grid = (M_TOT/MBLK, J_TOT/JBLK), block = 256 threads (8 wave32)
//   wave (wm,wj) in 4x2 grid computes a 32x64 sub-tile = 2x4 WMMA tiles.
// ---------------------------------------------------------------------------
__global__ __launch_bounds__(256) void qkv_gemm_kernel(
    const float* __restrict__ x,      // [M_TOT, DIM]
    const float* __restrict__ w,      // [J_TOT, DIM]  (qkv_w)
    const float* __restrict__ bias,   // [J_TOT]
    float* __restrict__ out_base)     // d_out; q/k/v live at chunks 1..3
{
    __shared__ __bf16 ldsA[MBLK * LDS_STRIDE];
    __shared__ __bf16 ldsB[JBLK * LDS_STRIDE];

    const int tid  = threadIdx.x;
    const int wave = tid >> 5;
    const int lane = tid & 31;

    const int m0 = blockIdx.x * MBLK;
    const int j0 = blockIdx.y * JBLK;

    const int wm = wave & 3;   // 0..3 -> 32-row group
    const int wj = wave >> 2;  // 0..1 -> 64-col group

    v8f acc[2][4];
#pragma unroll
    for (int i = 0; i < 2; ++i)
#pragma unroll
        for (int j = 0; j < 4; ++j)
            acc[i][j] = v8f{0.f, 0.f, 0.f, 0.f, 0.f, 0.f, 0.f, 0.f};

    // Staging: 256 threads x 16 contiguous fp32 = 128 rows x 32 cols per chunk.
    const int srow = tid >> 1;            // 0..127
    const int scol = (tid & 1) * 16;      // 0 or 16

    // Fragment addressing per ISA 16-bit A layout:
    // lane<16: row=lane,   K 0..7 then 16..23 ; lane>=16: row=lane-16, K 8..15 then 24..31
    const int frow = lane & 15;
    const int koff = (lane >> 4) * 8;

    for (int kc = 0; kc < DIM; kc += KBLK) {
        {   // ---- stage A chunk (x) ----
            const float* src = x + (size_t)(m0 + srow) * DIM + kc + scol;
            __bf16* dst = &ldsA[srow * LDS_STRIDE + scol];
#pragma unroll
            for (int i = 0; i < 16; i += 4) {
                v4f t = *(const v4f*)(src + i);
                dst[i + 0] = (__bf16)t.x;
                dst[i + 1] = (__bf16)t.y;
                dst[i + 2] = (__bf16)t.z;
                dst[i + 3] = (__bf16)t.w;
            }
            // ---- stage B chunk (qkv_w rows == B^T rows) ----
            const float* srcw = w + (size_t)(j0 + srow) * DIM + kc + scol;
            __bf16* dstw = &ldsB[srow * LDS_STRIDE + scol];
#pragma unroll
            for (int i = 0; i < 16; i += 4) {
                v4f t = *(const v4f*)(srcw + i);
                dstw[i + 0] = (__bf16)t.x;
                dstw[i + 1] = (__bf16)t.y;
                dstw[i + 2] = (__bf16)t.z;
                dstw[i + 3] = (__bf16)t.w;
            }
        }
        __syncthreads();

        v16bf afrag[2];
#pragma unroll
        for (int i = 0; i < 2; ++i) {
            const __bf16* p = &ldsA[(wm * 32 + i * 16 + frow) * LDS_STRIDE + koff];
            v8bf lo = *(const v8bf*)p;          // K 0..7  (or 8..15)
            v8bf hi = *(const v8bf*)(p + 16);   // K 16..23 (or 24..31)
            afrag[i] = __builtin_shufflevector(lo, hi,
                0, 1, 2, 3, 4, 5, 6, 7, 8, 9, 10, 11, 12, 13, 14, 15);
        }
        v16bf bfrag[4];
#pragma unroll
        for (int j = 0; j < 4; ++j) {
            const __bf16* p = &ldsB[(wj * 64 + j * 16 + frow) * LDS_STRIDE + koff];
            v8bf lo = *(const v8bf*)p;
            v8bf hi = *(const v8bf*)(p + 16);
            bfrag[j] = __builtin_shufflevector(lo, hi,
                0, 1, 2, 3, 4, 5, 6, 7, 8, 9, 10, 11, 12, 13, 14, 15);
        }

#pragma unroll
        for (int i = 0; i < 2; ++i)
#pragma unroll
            for (int j = 0; j < 4; ++j)
                acc[i][j] = __builtin_amdgcn_wmma_f32_16x16x32_bf16(
                    /*neg_a=*/false, afrag[i],
                    /*neg_b=*/false, bfrag[j],
                    /*c_mod=*/(short)0, acc[i][j],
                    /*reuse_a=*/false, /*reuse_b=*/false);

        __syncthreads();
    }

    // ---- epilogue: bias add + scatter to q/k/v [N,H,B,Hd] layout ----
    // C/D layout: VGPR r, lane L -> row = r + 8*(L>=16), col = L%16
#pragma unroll
    for (int i = 0; i < 2; ++i) {
        const int mbase = m0 + wm * 32 + i * 16;
        const int bb    = mbase / SEQ;                       // batch index (tile-uniform)
        const int nbase = (mbase % SEQ) + (lane >> 4) * 8;   // seq index base for this lane
#pragma unroll
        for (int jf = 0; jf < 4; ++jf) {
            const int jcol  = j0 + wj * 64 + jf * 16 + (lane & 15);
            const int which = jcol >> 9;   // 0=q, 1=k, 2=v
            const int dcol  = jcol & 511;
            const int h     = dcol >> 6;
            const int hd    = dcol & 63;
            const float bj  = bias[jcol];
            float* dst = out_base + OUT_CHUNK * (size_t)(1 + which)
                       + (size_t)h * (BATCH * HDIM) + (size_t)bb * HDIM + hd;
#pragma unroll
            for (int r = 0; r < 8; ++r) {
                dst[(size_t)(nbase + r) * (HEADS * BATCH * HDIM)] = acc[i][jf][r] + bj;
            }
        }
    }
}

// ---------------------------------------------------------------------------
// Kernel 2: Vsum[f] = sum over all (n,h,b) of v[n,h,b,f]   (262144 rows x 64)
//   grid = 256 blocks x 256 threads; atomicAdd partials into vsum (memset'd).
// ---------------------------------------------------------------------------
__global__ __launch_bounds__(256) void vsum_kernel(
    const float* __restrict__ v, float* __restrict__ vsum)
{
    __shared__ float red[256];
    const int t  = threadIdx.x;
    const int f  = t & 63;
    const int rg = t >> 6;               // 0..3
    const int RPB = (SEQ * HEADS * BATCH) / 256;  // 1024 rows per block
    const int r0 = blockIdx.x * RPB;

    float s = 0.f;
    for (int r = r0 + rg; r < r0 + RPB; r += 4)
        s += v[(size_t)r * HDIM + f];

    red[t] = s;
    __syncthreads();
    if (rg == 0) {
        float tot = red[f] + red[64 + f] + red[128 + f] + red[192 + f];
        atomicAdd(&vsum[f], tot);
    }
}

// ---------------------------------------------------------------------------
// Kernel 3: outvec[j] = sum_d proj_w[j,d] * Vsum[d%64] + proj_b[j]
//   1 block x 512 threads.
// ---------------------------------------------------------------------------
__global__ __launch_bounds__(512) void outvec_kernel(
    const float* __restrict__ proj_w, const float* __restrict__ proj_b,
    const float* __restrict__ vsum, float* __restrict__ outvec)
{
    __shared__ float vs[DIM];
    const int t = threadIdx.x;           // 0..511
    vs[t] = vsum[t & 63];
    __syncthreads();

    float s = proj_b[t];
    const float* wr = proj_w + (size_t)t * DIM;
    for (int d = 0; d < DIM; d += 4) {
        v4f w4 = *(const v4f*)(wr + d);
        s += w4.x * vs[d] + w4.y * vs[d + 1] + w4.z * vs[d + 2] + w4.w * vs[d + 3];
    }
    outvec[t] = s;
}

// ---------------------------------------------------------------------------
// Kernel 4: broadcast outvec (512 floats) across out[B*N, 512] (64 MB, float4)
// ---------------------------------------------------------------------------
__global__ __launch_bounds__(256) void bcast_kernel(
    const float* __restrict__ outvec, float* __restrict__ out)
{
    __shared__ float ov[DIM];
    for (int i = threadIdx.x; i < DIM; i += blockDim.x) ov[i] = outvec[i];
    __syncthreads();

    const size_t n4     = OUT_CHUNK / 4;       // float4 elements
    const size_t stride = (size_t)gridDim.x * blockDim.x;
    for (size_t i = (size_t)blockIdx.x * blockDim.x + threadIdx.x; i < n4; i += stride) {
        const int col4 = (int)(i & 127);       // 128 float4 per 512-float row
        v4f val = *(const v4f*)(ov + col4 * 4);
        ((v4f*)out)[i] = val;
    }
}

// ---------------------------------------------------------------------------
// Host-side launcher
// ---------------------------------------------------------------------------
extern "C" void kernel_launch(void* const* d_in, const int* in_sizes, int n_in,
                              void* d_out, int out_size, void* d_ws, size_t ws_size,
                              hipStream_t stream) {
    const float* x      = (const float*)d_in[0];
    const float* qkv_w  = (const float*)d_in[1];
    const float* qkv_b  = (const float*)d_in[2];
    const float* proj_w = (const float*)d_in[3];
    const float* proj_b = (const float*)d_in[4];

    float* out    = (float*)d_out;
    float* vsum   = (float*)d_ws;        // 64 floats
    float* outvec = vsum + 64;           // 512 floats

    // 1) QKV projection GEMM -> writes q,k,v (chunks 1..3 of d_out)
    dim3 gA(M_TOT / MBLK, J_TOT / JBLK);
    qkv_gemm_kernel<<<gA, 256, 0, stream>>>(x, qkv_w, qkv_b, out);

    // 2) Vsum over v (chunk 3)
    hipMemsetAsync(vsum, 0, 64 * sizeof(float), stream);
    vsum_kernel<<<256, 256, 0, stream>>>(out + 3 * OUT_CHUNK, vsum);

    // 3) single output row
    outvec_kernel<<<1, 512, 0, stream>>>(proj_w, proj_b, vsum, outvec);

    // 4) broadcast to out (chunk 0)
    bcast_kernel<<<1024, 256, 0, stream>>>(outvec, out);
}
